// CustomLSTM_29798483100220
// MI455X (gfx1250) — matrix operand order repaired and checked by reading, compiled, not run on hardware
//
#include <hip/hip_runtime.h>
#include <hip/hip_bf16.h>
#include <math.h>

// ---- sizes ----
#define B_  64
#define S_  1024
#define I_  512
#define H_  512
#define G_  2048   // 4*H
#define K_  1024   // I + H

// ---- trivial vector types (safe inside unions, lower to b128 loads) ----
typedef __bf16        v16bf __attribute__((ext_vector_type(16)));
typedef float         v8f   __attribute__((ext_vector_type(8)));
typedef unsigned int  u32x4 __attribute__((ext_vector_type(4)));
typedef float         f32x4 __attribute__((ext_vector_type(4)));

union Frag {
    v16bf v;
    u32x4 q[2];
    unsigned u[8];
};

__device__ __forceinline__ unsigned pack2_bf16(float lo, float hi) {
    union { float f; unsigned u; } a, b;
    a.f = lo; b.f = hi;
    unsigned rl = a.u + 0x7fffu + ((a.u >> 16) & 1u);   // round-to-nearest-even
    unsigned rh = b.u + 0x7fffu + ((b.u >> 16) & 1u);
    return (rl >> 16) | (rh & 0xffff0000u);
}

__device__ __forceinline__ unsigned short f32_to_bf16(float f) {
    union { float f; unsigned u; } a; a.f = f;
    unsigned r = a.u + 0x7fffu + ((a.u >> 16) & 1u);
    return (unsigned short)(r >> 16);
}

// -----------------------------------------------------------------------------
// One-time: build combined bf16 weight matrix Wc[N=2048][K=1024] (K contiguous
// per output column N), K = [ W rows (512) | U rows (512) ].
// -----------------------------------------------------------------------------
__global__ __launch_bounds__(256)
void lstm_prep(const float* __restrict__ W, const float* __restrict__ U,
               unsigned short* __restrict__ wc) {
    const int n = blockIdx.x;                 // 0..2047 output column
    const int t = threadIdx.x;                // 0..255
    unsigned short* row = wc + (size_t)n * K_;
    #pragma unroll
    for (int j = 0; j < 4; ++j) {
        int k = t * 4 + j;                    // 0..1023
        float f = (k < I_) ? W[(size_t)k * G_ + n]
                           : U[(size_t)(k - I_) * G_ + n];
        row[k] = f32_to_bf16(f);
    }
}

// -----------------------------------------------------------------------------
// One-time: convert the full activation tensor x (f32) to bf16 so the step
// kernels do pure b128 loads (no per-step conversion VALU, no x128 redundant
// conversion work). 8 elements / thread, fully vectorized.
// -----------------------------------------------------------------------------
__global__ __launch_bounds__(256)
void convert_x(const float* __restrict__ x, unsigned short* __restrict__ xb) {
    const size_t i = (size_t)blockIdx.x * 256 + threadIdx.x;  // 8 elems each
    const f32x4* src = (const f32x4*)x + i * 2;
    f32x4 a = src[0], b = src[1];
    u32x4 o;
    o.x = pack2_bf16(a.x, a.y); o.y = pack2_bf16(a.z, a.w);
    o.z = pack2_bf16(b.x, b.y); o.w = pack2_bf16(b.z, b.w);
    ((u32x4*)xb)[i] = o;
}

// -----------------------------------------------------------------------------
// One timestep: gates = [x_t | h_prev] @ [W;U] + bias  (K=1024, bf16 WMMA,
// two accumulator chains to halve WMMA RAW latency), then fused sigmoid/tanh
// epilogue updating c, writing h (f32 to out, bf16 ping-pong for next step).
// grid = (32 hidden 16-col blocks, 4 row tiles of 16), block = 128 (4 waves,
// one wave per gate i/f/g/o).
// XPRE: x pre-converted to bf16 in workspace (preferred) vs f32 + in-register
// conversion fallback when ws is too small.
// -----------------------------------------------------------------------------
template <bool XPRE>
__global__ __launch_bounds__(128)
void lstm_step(const float* __restrict__ x,                // [64][1024][512] f32
               const unsigned short* __restrict__ xbf,     // same, bf16 (XPRE)
               const unsigned short* __restrict__ wc,      // [2048][1024] bf16
               const float* __restrict__ bias,             // [2048]
               const unsigned short* __restrict__ h_prev,  // [64][512] bf16
               unsigned short* __restrict__ h_next,        // [64][512] bf16
               float* __restrict__ c,                      // [64][512] f32
               float* __restrict__ out,                    // hs + h_t + c_t
               int t) {
    __shared__ float gbuf[4][16][16];

    const int lane = threadIdx.x & 31;
    const int gate = threadIdx.x >> 5;        // wave id == gate 0..3 (i,f,g,o)
    const int m    = blockIdx.y;              // row tile (16 batch rows)
    const int hb   = blockIdx.x * 16;         // hidden column base

    const int colN = lane & 15;
    const int half = lane >> 4;

    const int arow = m * 16 + colN;           // A-matrix row (batch index)
    const int ncol = gate * H_ + hb + colN;   // B-matrix column (gate column)

    const unsigned short* hrow = h_prev + (size_t)arow * H_;
    const unsigned short* wrow = wc + (size_t)ncol * K_;

    v8f acc0 = {}, acc1 = {};

    // ---- K-blocks 0..15: A from x_t ----
    if (XPRE) {
        const unsigned short* xr = xbf + ((size_t)arow * S_ + t) * I_;
        for (int kb = 0; kb < 16; kb += 2) {
            #pragma unroll
            for (int u = 0; u < 2; ++u) {
                const int kk = (kb + u) * 32;
                Frag a, b;
                a.q[0] = *(const u32x4*)(xr + kk + 8 * half);
                a.q[1] = *(const u32x4*)(xr + kk + 16 + 8 * half);
                const u32x4* wp = (const u32x4*)(wrow + kk + 8 * half);
                b.q[0] = wp[0]; b.q[1] = wp[2];
                if (u == 0)
                    acc0 = __builtin_amdgcn_wmma_f32_16x16x32_bf16(
                        false, a.v, false, b.v, (short)0, acc0, false, false);
                else
                    acc1 = __builtin_amdgcn_wmma_f32_16x16x32_bf16(
                        false, a.v, false, b.v, (short)0, acc1, false, false);
            }
        }
    } else {
        const float* xrow = x + ((size_t)arow * S_ + t) * I_;
        for (int kb = 0; kb < 16; kb += 2) {
            #pragma unroll
            for (int u = 0; u < 2; ++u) {
                const int kk = (kb + u) * 32;
                Frag a, b;
                const f32x4* p0 = (const f32x4*)(xrow + kk + 8 * half);
                const f32x4* p1 = (const f32x4*)(xrow + kk + 16 + 8 * half);
                f32x4 f0 = p0[0], f1 = p0[1];
                f32x4 f2 = p1[0], f3 = p1[1];
                a.u[0] = pack2_bf16(f0.x, f0.y); a.u[1] = pack2_bf16(f0.z, f0.w);
                a.u[2] = pack2_bf16(f1.x, f1.y); a.u[3] = pack2_bf16(f1.z, f1.w);
                a.u[4] = pack2_bf16(f2.x, f2.y); a.u[5] = pack2_bf16(f2.z, f2.w);
                a.u[6] = pack2_bf16(f3.x, f3.y); a.u[7] = pack2_bf16(f3.z, f3.w);
                const u32x4* wp = (const u32x4*)(wrow + kk + 8 * half);
                b.q[0] = wp[0]; b.q[1] = wp[2];
                if (u == 0)
                    acc0 = __builtin_amdgcn_wmma_f32_16x16x32_bf16(
                        false, a.v, false, b.v, (short)0, acc0, false, false);
                else
                    acc1 = __builtin_amdgcn_wmma_f32_16x16x32_bf16(
                        false, a.v, false, b.v, (short)0, acc1, false, false);
            }
        }
    }

    // ---- K-blocks 16..31: A from h_prev (bf16: pure b128 loads) ----
    for (int kb = 0; kb < 16; kb += 2) {
        #pragma unroll
        for (int u = 0; u < 2; ++u) {
            const int kk = (kb + u) * 32;
            Frag a, b;
            a.q[0] = *(const u32x4*)(hrow + kk + 8 * half);
            a.q[1] = *(const u32x4*)(hrow + kk + 16 + 8 * half);
            const u32x4* wp = (const u32x4*)(wrow + I_ + kk + 8 * half);
            b.q[0] = wp[0]; b.q[1] = wp[2];
            if (u == 0)
                acc0 = __builtin_amdgcn_wmma_f32_16x16x32_bf16(
                    false, a.v, false, b.v, (short)0, acc0, false, false);
            else
                acc1 = __builtin_amdgcn_wmma_f32_16x16x32_bf16(
                    false, a.v, false, b.v, (short)0, acc1, false, false);
        }
    }

    v8f acc = acc0 + acc1;

    // ---- stash gate tile: lane holds N=colN, M = r + 8*half ----
    #pragma unroll
    for (int r = 0; r < 8; ++r)
        gbuf[gate][r + 8 * half][colN] = acc[r];

    __syncthreads();

    // ---- fused LSTM epilogue over this block's 16x16 hidden patch ----
    #pragma unroll
    for (int j = 0; j < 2; ++j) {
        const int e   = threadIdx.x + 128 * j;  // 0..255
        const int r16 = e >> 4, col = e & 15;
        const int row = m * 16 + r16;           // batch index
        const int hc  = hb + col;               // hidden index

        float gi = gbuf[0][r16][col] + bias[0 * H_ + hc];
        float gf = gbuf[1][r16][col] + bias[1 * H_ + hc];
        float gg = gbuf[2][r16][col] + bias[2 * H_ + hc];
        float go = gbuf[3][r16][col] + bias[3 * H_ + hc];

        float i_ = 1.0f / (1.0f + __expf(-gi));
        float f_ = 1.0f / (1.0f + __expf(-gf));
        float g_ = tanhf(gg);
        float o_ = 1.0f / (1.0f + __expf(-go));

        const size_t ci = (size_t)row * H_ + hc;
        float cn = f_ * c[ci] + i_ * g_;
        c[ci] = cn;
        float hn = o_ * tanhf(cn);

        out[((size_t)row * S_ + t) * H_ + hc] = hn;   // hidden_seq
        h_next[ci] = f32_to_bf16(hn);                 // recurrent input for t+1

        if (t == S_ - 1) {
            const size_t tail = (size_t)B_ * S_ * H_;
            out[tail + ci]                   = hn;    // h_t
            out[tail + (size_t)B_ * H_ + ci] = cn;    // c_t
        }
    }
}

// -----------------------------------------------------------------------------
// Host-side launch. All graph-capture safe (kernels + hipMemsetAsync only).
// -----------------------------------------------------------------------------
extern "C" void kernel_launch(void* const* d_in, const int* in_sizes, int n_in,
                              void* d_out, int out_size, void* d_ws, size_t ws_size,
                              hipStream_t stream) {
    const float* x    = (const float*)d_in[0];
    const float* W    = (const float*)d_in[1];
    const float* U    = (const float*)d_in[2];
    const float* bias = (const float*)d_in[3];
    float* out = (float*)d_out;

    char* ws = (char*)d_ws;
    const size_t WC_BYTES  = (size_t)G_ * K_ * sizeof(unsigned short);       // 4 MB
    const size_t XB_BYTES  = (size_t)B_ * S_ * I_ * sizeof(unsigned short);  // 64 MB
    const size_t HB_BYTES  = (size_t)B_ * H_ * sizeof(unsigned short);       // 64 KB
    const size_t C_BYTES   = (size_t)B_ * H_ * sizeof(float);                // 128 KB
    const size_t NEED_PRE  = WC_BYTES + XB_BYTES + 2 * HB_BYTES + C_BYTES;
    (void)in_sizes; (void)n_in; (void)out_size;

    const bool xpre = (ws_size >= NEED_PRE);

    unsigned short* wc = (unsigned short*)ws;
    char* p = ws + WC_BYTES;
    unsigned short* xbf = nullptr;
    if (xpre) { xbf = (unsigned short*)p; p += XB_BYTES; }
    unsigned short* hb0 = (unsigned short*)p;
    unsigned short* hb1 = (unsigned short*)(p + HB_BYTES);
    float*          c   = (float*)(p + 2 * HB_BYTES);

    // bf16 weight pack + (optionally) bf16 activation pack
    lstm_prep<<<dim3(G_), dim3(256), 0, stream>>>(W, U, wc);
    if (xpre) {
        const size_t n8 = (size_t)B_ * S_ * I_ / 8;          // 8 elems / thread
        convert_x<<<dim3((unsigned)(n8 / 256)), dim3(256), 0, stream>>>(x, xbf);
    }
    // zero initial h, c
    hipMemsetAsync(hb0, 0, HB_BYTES, stream);
    hipMemsetAsync(c,   0, C_BYTES, stream);

    for (int t = 0; t < S_; ++t) {
        const unsigned short* hp = (t & 1) ? hb1 : hb0;
        unsigned short*       hn = (t & 1) ? hb0 : hb1;
        if (xpre)
            lstm_step<true><<<dim3(H_ / 16, B_ / 16), dim3(128), 0, stream>>>(
                x, xbf, wc, bias, hp, hn, c, out, t);
        else
            lstm_step<false><<<dim3(H_ / 16, B_ / 16), dim3(128), 0, stream>>>(
                x, xbf, wc, bias, hp, hn, c, out, t);
    }
}
